// TransformerBlock_90898687852803
// MI455X (gfx1250) — matrix operand order repaired and checked by reading, not compiled, not run on hardware
//
#include <hip/hip_runtime.h>
#include <hip/hip_bf16.h>
#include <math.h>

// ---- problem constants (match reference) ----
#define NTOK 9216      // N = D*H*W = 16*24*24, interpreted as (x,y,z)=(24,24,16), n = x*384+y*16+z
#define CCH  128
#define NHEADS 8
#define HDIM 16
#define PLN 144        // pooled tokens
#define LLW 27         // 3x3x3 window
#define TTBL 4096
#define NEGV -1e30f

typedef __attribute__((ext_vector_type(16))) __bf16 v16bf;
typedef __attribute__((ext_vector_type(8)))  __bf16 v8bf;
typedef __attribute__((ext_vector_type(8)))  float  v8f;

union ABFrag { v16bf v; v8bf h[2]; __bf16 e[16]; };

__device__ __forceinline__ float geluf(float x)     { return 0.5f * x * (1.0f + erff(x * 0.70710678118654752f)); }
__device__ __forceinline__ float softplusf(float x) { return (x > 20.f) ? x : log1pf(expf(x)); }

// ---- CDNA5 async global->LDS copy (ASYNCcnt-tracked DMA into LDS) ----
// VDST = 32-bit LDS byte offset (low 32 bits of the shared-aperture address),
// VADDR = 64-bit global address, saddr = off.  See CDNA5 ISA.
__device__ __forceinline__ void async_copy_b128(void* lds_dst, const void* gsrc) {
  unsigned loff = (unsigned)(size_t)lds_dst;
  asm volatile("global_load_async_to_lds_b128 %0, %1, off" :: "v"(loff), "v"(gsrc) : "memory");
}
__device__ __forceinline__ void wait_async() {
  asm volatile("s_wait_asynccnt 0" ::: "memory");
}

// =====================================================================
// Elementwise / prep kernels
// =====================================================================
__global__ __launch_bounds__(256) void k_transpose_in(const float* __restrict__ x, float* __restrict__ xs) {
  int i = blockIdx.x * 256 + threadIdx.x;           // i = s*128 + c
  if (i >= NTOK * CCH) return;
  int s = i >> 7, c = i & 127;
  xs[i] = x[c * NTOK + s] + 1.0f;
}

__global__ __launch_bounds__(256) void k_f2bf(const float* __restrict__ in, __bf16* __restrict__ out, int count) {
  int i = blockIdx.x * 256 + threadIdx.x;
  if (i < count) out[i] = (__bf16)in[i];
}

// pack conv weight (O,I,3,3,3) -> wpk[tap][o][i] bf16
__global__ __launch_bounds__(256) void k_pack_convw(const float* __restrict__ w, __bf16* __restrict__ wpk) {
  int i = blockIdx.x * 256 + threadIdx.x;
  if (i >= 27 * CCH * CCH) return;
  int tap = i / (CCH * CCH);
  int o   = (i / CCH) % CCH;
  int ci  = i % CCH;
  wpk[i] = (__bf16)w[o * (CCH * 27) + ci * 27 + tap];
}

// LayerNorm over C=128 per row, bf16 output
__global__ __launch_bounds__(128) void k_layernorm(const float* __restrict__ in, const float* __restrict__ w,
                                                   const float* __restrict__ b, __bf16* __restrict__ out) {
  __shared__ float red[128];
  int r = blockIdx.x, t = threadIdx.x;
  float xv = in[r * 128 + t];
  red[t] = xv; __syncthreads();
  for (int s = 64; s > 0; s >>= 1) { if (t < s) red[t] += red[t + s]; __syncthreads(); }
  float m = red[0] * (1.f / 128.f); __syncthreads();
  float d = xv - m;
  red[t] = d * d; __syncthreads();
  for (int s = 64; s > 0; s >>= 1) { if (t < s) red[t] += red[t + s]; __syncthreads(); }
  float v = red[0] * (1.f / 128.f);
  out[r * 128 + t] = (__bf16)(d * rsqrtf(v + 1e-5f) * w[t] + b[t]);
}

// =====================================================================
// WMMA GEMM: out(M,Nc) = A(M,128) * W(Nc,128)^T (+bias)(+gelu)
// one block per 16-row M tile; A tile async-staged to LDS once, each wave
// keeps all 4 K-fragments of A in registers and loops over N tiles.
// All 4 B fragments are preloaded (one clause) before the WMMA chain so the
// VMEM latency overlaps the matrix pipe.
// =====================================================================
__global__ __launch_bounds__(256) void k_gemm128(const __bf16* __restrict__ A, const __bf16* __restrict__ W,
                                                 const float* __restrict__ bias, float* __restrict__ out,
                                                 int ntiles, int act) {
  __shared__ __bf16 sA[16 * 128];                    // 4 KB
  int wave = threadIdx.x >> 5, lane = threadIdx.x & 31;
  int tm = blockIdx.x;
  {                                                  // 256 lanes x 16B = whole tile
    int row = threadIdx.x >> 4, seg = threadIdx.x & 15;
    async_copy_b128(sA + row * 128 + seg * 8, A + (size_t)(tm * 16 + row) * 128 + seg * 8);
  }
  wait_async();
  __syncthreads();
  int lhalf = (lane < 16) ? 0 : 1;
  const __bf16* arow = sA + (lane & 15) * 128 + lhalf * 16;
  ABFrag fr[4];
#pragma unroll
  for (int k4 = 0; k4 < 4; ++k4) {
    fr[k4].h[0] = *(const v8bf*)(arow + k4 * 32);
    fr[k4].h[1] = *(const v8bf*)(arow + k4 * 32 + 8);
  }
  int Nc = ntiles * 16;
  for (int tn = wave; tn < ntiles; tn += 8) {        // wave-uniform bound: EXEC all-ones for WMMA
    const __bf16* wrow = W + (size_t)(tn * 16 + (lane & 15)) * 128 + lhalf * 16;
    ABFrag b[4];
#pragma unroll
    for (int k4 = 0; k4 < 4; ++k4) b[k4].v = *(const v16bf*)(wrow + k4 * 32);
    v8f acc = {0.f, 0.f, 0.f, 0.f, 0.f, 0.f, 0.f, 0.f};
#pragma unroll
    for (int k4 = 0; k4 < 4; ++k4)
      acc = __builtin_amdgcn_wmma_f32_16x16x32_bf16(false, fr[k4].v, false, b[k4].v, (short)0, acc, false, false);
    int ncol = tn * 16 + (lane & 15);
    float bv = bias ? bias[ncol] : 0.f;
#pragma unroll
    for (int vv = 0; vv < 8; ++vv) {
      int m = tm * 16 + vv + lhalf * 8;
      float val = acc[vv] + bv;
      if (act == 1) val = geluf(val);
      out[(size_t)m * Nc + ncol] = val;
    }
  }
}

// =====================================================================
// 3x3x3 conv as implicit GEMM over z-columns (D=16 == M-tile), WMMA.
// One block per (x,y) column; the 3x3 neighborhood of z-columns (9 x 16 x 128
// bf16 = 36 KB) is async-staged into LDS once and shared by all 8 co-tiles.
// Per tap: preload 4 A frags (ds) + 4 B frags (global) then 4 WMMAs.
// spatial layout: n = (x*24+y)*16+z ; act (N,128) bf16 ; wpk[tap][o][i]
// =====================================================================
__global__ __launch_bounds__(256) void k_conv3d_wmma(const __bf16* __restrict__ act, const __bf16* __restrict__ wpk,
                                                     const float* __restrict__ bias, float* __restrict__ out) {
  __shared__ __bf16 sact[9 * 16 * 128];              // 36,864 B
  int wave = threadIdx.x >> 5, lane = threadIdx.x & 31;
  int col = blockIdx.x;                              // 576 (x,y) columns
  int x = col / 24, y = col % 24;
  {
    int row = threadIdx.x >> 4, seg = threadIdx.x & 15;
    for (int ci = 0; ci < 9; ++ci) {                 // block-uniform validity
      int xx = x + ci / 3 - 1, yy = y + ci % 3 - 1;
      if (xx >= 0 && xx < 24 && yy >= 0 && yy < 24) {
        async_copy_b128(sact + (ci * 16 + row) * 128 + seg * 8,
                        act + ((size_t)((xx * 24 + yy) * 16 + row)) * 128 + seg * 8);
      }
    }
  }
  wait_async();
  __syncthreads();
  int cot = wave;                                    // co tile 0..7
  int lhalf = (lane < 16) ? 0 : 1;
  int zr = lane & 15;
  const __bf16* wbase = wpk + (size_t)(cot * 16 + (lane & 15)) * 128 + lhalf * 16;
  v8f acc = {0.f, 0.f, 0.f, 0.f, 0.f, 0.f, 0.f, 0.f};
  for (int tap = 0; tap < 27; ++tap) {
    int dx = tap / 9 - 1, dy = (tap / 3) % 3 - 1, dz = tap % 3 - 1;
    int ci = (dx + 1) * 3 + (dy + 1);
    int xx = x + dx, yy = y + dy, zz = zr + dz;
    bool valid = (xx >= 0) && (xx < 24) && (yy >= 0) && (yy < 24) && (zz >= 0) && (zz < 16);
    int zc = valid ? zz : 0;
    const __bf16* arow = sact + (ci * 16 + zc) * 128 + lhalf * 16;
    const __bf16* wrow = wbase + (size_t)tap * (128 * 128);
    if (tap + 1 < 27) __builtin_prefetch(wbase + (size_t)(tap + 1) * (128 * 128), 0, 0);
    ABFrag a[4], b[4];
#pragma unroll
    for (int k4 = 0; k4 < 4; ++k4) {
      if (valid) {
        a[k4].h[0] = *(const v8bf*)(arow + k4 * 32);
        a[k4].h[1] = *(const v8bf*)(arow + k4 * 32 + 8);
      } else {
#pragma unroll
        for (int q = 0; q < 16; ++q) a[k4].e[q] = (__bf16)0.0f;
      }
      b[k4].v = *(const v16bf*)(wrow + k4 * 32);
    }
#pragma unroll
    for (int k4 = 0; k4 < 4; ++k4)
      acc = __builtin_amdgcn_wmma_f32_16x16x32_bf16(false, a[k4].v, false, b[k4].v, (short)0, acc, false, false);
  }
  int ncol = cot * 16 + (lane & 15);
  float bv = bias[ncol];
#pragma unroll
  for (int vv = 0; vv < 8; ++vv) {
    int n = (x * 24 + y) * 16 + vv + lhalf * 8;
    out[(size_t)n * 128 + ncol] = acc[vv] + bv;
  }
}

// =====================================================================
// q/k normalization (in-place in qkvv), q_scaled + q_norm
// =====================================================================
__global__ __launch_bounds__(256) void k_qk_norm(float* __restrict__ qkvv, float* __restrict__ qnorm,
                                                 const float* __restrict__ qe, const float* __restrict__ temp,
                                                 const float* __restrict__ sls) {
  int u = blockIdx.x * 256 + threadIdx.x;
  if (u >= NTOK * NHEADS) return;
  int n = u >> 3, h = u & 7;
  float* qp = qkvv + (size_t)n * 512 + h * 16;
  float* kp = qp + 128;
  float nq = 0.f, nk = 0.f;
  for (int d = 0; d < 16; ++d) { nq += qp[d] * qp[d]; nk += kp[d] * kp[d]; }
  nq = fmaxf(sqrtf(nq), 1e-12f);
  nk = fmaxf(sqrtf(nk), 1e-12f);
  float sp = softplusf(temp[h]) * sls[0];
  for (int d = 0; d < 16; ++d) {
    float qn = qp[d] / nq;
    qnorm[(size_t)n * 128 + h * 16 + d] = qn;
    qp[d] = (qn + qe[h * 16 + d]) * sp;
    kp[d] = kp[d] / nk;
  }
}

// 4x4x4 average pool of srout (N,128) -> xp (PL,128)
__global__ __launch_bounds__(128) void k_pool(const float* __restrict__ srout, float* __restrict__ xp) {
  int p = blockIdx.x, c = threadIdx.x;
  int px = p / 24, py = (p / 4) % 6, pz = p % 4;
  float s = 0.f;
  for (int ix = 0; ix < 4; ++ix)
    for (int iy = 0; iy < 4; ++iy)
      for (int iz = 0; iz < 4; ++iz) {
        int n = ((px * 4 + ix) * 24 + (py * 4 + iy)) * 16 + (pz * 4 + iz);
        s += srout[(size_t)n * 128 + c];
      }
  xp[p * 128 + c] = s * (1.f / 64.f);
}

// normalize k_pool, copy v_pool into [h][p][d] layout
__global__ __launch_bounds__(256) void k_norm_kpool(const float* __restrict__ kvp, float* __restrict__ kpooln,
                                                    float* __restrict__ vpool) {
  int u = blockIdx.x * 256 + threadIdx.x;
  if (u >= NHEADS * PLN) return;
  int h = u / PLN, p = u % PLN;
  const float* kr = kvp + p * 256 + h * 16;
  const float* vr = kvp + p * 256 + 128 + h * 16;
  float s = 0.f;
  for (int d = 0; d < 16; ++d) s += kr[d] * kr[d];
  float inv = 1.f / fmaxf(sqrtf(s), 1e-12f);
  for (int d = 0; d < 16; ++d) {
    kpooln[(h * PLN + p) * 16 + d] = kr[d] * inv;
    vpool[(h * PLN + p) * 16 + d] = vr[d];
  }
}

// CPB MLP: tb[r][h] = relu(rct@w1.T+b1) @ w2.T + b2
__global__ __launch_bounds__(256) void k_cpb(const float* __restrict__ rct, const float* __restrict__ w1,
                                             const float* __restrict__ b1, const float* __restrict__ w2,
                                             const float* __restrict__ b2, float* __restrict__ tb) {
  int r = blockIdx.x * 256 + threadIdx.x;
  if (r >= TTBL) return;
  float c0 = rct[r * 3], c1 = rct[r * 3 + 1], c2 = rct[r * 3 + 2];
  float o[8];
  for (int h = 0; h < 8; ++h) o[h] = b2[h];
  for (int j = 0; j < 512; ++j) {
    float hj = c0 * w1[j * 3] + c1 * w1[j * 3 + 1] + c2 * w1[j * 3 + 2] + b1[j];
    hj = fmaxf(hj, 0.f);
    for (int h = 0; h < 8; ++h) o[h] += hj * w2[h * 512 + j];
  }
  for (int h = 0; h < 8; ++h) tb[r * 8 + h] = o[h];
}

// =====================================================================
// Fused local-window + pooled attention with joint softmax (VALU)
// grid (N/64, HEADS), block 64 ; 62 KB LDS/workgroup
// =====================================================================
__global__ __launch_bounds__(64) void k_attn(const float* __restrict__ qkvv, const float* __restrict__ qnorm,
                                             const float* __restrict__ kpooln, const float* __restrict__ vpool,
                                             const float* __restrict__ tb, const int* __restrict__ rpi,
                                             const float* __restrict__ rpb, const float* __restrict__ ltok,
                                             const float* __restrict__ lbias, float* __restrict__ xSA) {
  __shared__ float s_kp[PLN * 16];
  __shared__ float s_vp[PLN * 16];
  __shared__ float s_lg[64 * 172];
  int h = blockIdx.y, t = threadIdx.x;
  for (int i = t; i < PLN * 16; i += 64) {
    s_kp[i] = kpooln[h * PLN * 16 + i];
    s_vp[i] = vpool[h * PLN * 16 + i];
  }
  __syncthreads();
  int n = blockIdx.x * 64 + t;
  int x = n / 384, y = (n % 384) >> 4, z = n & 15;
  float q[16], qn[16];
  const float* qp = qkvv + (size_t)n * 512 + h * 16;
  for (int d = 0; d < 16; ++d) { q[d] = qp[d]; qn[d] = qnorm[(size_t)n * 128 + h * 16 + d]; }
  float* lg = s_lg + t * 172;
  float mx = -3.4e38f;
  // 27 local logits
  for (int l = 0; l < 27; ++l) {
    int xx = x + l / 9 - 1, yy = y + (l / 3) % 3 - 1, zz = z + l % 3 - 1;
    float v;
    if (xx >= 0 && xx < 24 && yy >= 0 && yy < 24 && zz >= 0 && zz < 16) {
      const float* kr = qkvv + (size_t)((xx * 24 + yy) * 16 + zz) * 512 + 128 + h * 16;
      float s = 0.f;
      for (int d = 0; d < 16; ++d) s += q[d] * kr[d];
      v = s + rpb[h * 27 + l];
    } else v = NEGV;
    lg[l] = v; mx = fmaxf(mx, v);
  }
  // 144 pooled logits
  const int* rp = rpi + (size_t)n * PLN;
  for (int p = 0; p < PLN; ++p) {
    const float* kr = s_kp + p * 16;
    float s = 0.f;
    for (int d = 0; d < 16; ++d) s += q[d] * kr[d];
    float v = s + tb[rp[p] * 8 + h];
    lg[27 + p] = v; mx = fmaxf(mx, v);
  }
  float sum = 0.f;
  for (int i = 0; i < 171; ++i) { float e = expf(lg[i] - mx); lg[i] = e; sum += e; }
  float inv = 1.f / sum;
  float acc[16];
  for (int d = 0; d < 16; ++d) acc[d] = 0.f;
  // local output (learnable-token term applies to all l; padded v == 0)
  for (int l = 0; l < 27; ++l) {
    int xx = x + l / 9 - 1, yy = y + (l / 3) % 3 - 1, zz = z + l % 3 - 1;
    bool valid = xx >= 0 && xx < 24 && yy >= 0 && yy < 24 && zz >= 0 && zz < 16;
    float extra = lbias[h * 27 + l];
    for (int d = 0; d < 16; ++d) extra += qn[d] * ltok[(h * 16 + d) * 27 + l];
    float a = lg[l] * inv + extra;
    if (valid) {
      const float* vr = qkvv + (size_t)((xx * 24 + yy) * 16 + zz) * 512 + 384 + h * 16;
      for (int d = 0; d < 16; ++d) acc[d] += a * vr[d];
    }
  }
  for (int p = 0; p < PLN; ++p) {
    float a = lg[27 + p] * inv;
    const float* vr = s_vp + p * 16;
    for (int d = 0; d < 16; ++d) acc[d] += a * vr[d];
  }
  for (int d = 0; d < 16; ++d) xSA[(size_t)n * 128 + h * 16 + d] = acc[d];
}

// =====================================================================
// Channel attention (uses RAW q,k before in-place normalization!)
// =====================================================================
__global__ __launch_bounds__(256) void k_ca_gram(const float* __restrict__ qkvv, float* __restrict__ graw,
                                                 float* __restrict__ sq2, float* __restrict__ sk2) {
  __shared__ float sq[128][16];
  __shared__ float sk[128][16];
  int h = blockIdx.x, t = threadIdx.x;
  int d = t >> 4, e = t & 15;
  float g = 0.f, accq = 0.f, acck = 0.f;
  for (int base = 0; base < NTOK; base += 128) {
    for (int i = t; i < 128 * 16; i += 256) {
      int n = base + (i >> 4), dd = i & 15;
      sq[i >> 4][dd] = qkvv[(size_t)n * 512 + h * 16 + dd];
      sk[i >> 4][dd] = qkvv[(size_t)n * 512 + 128 + h * 16 + dd];
    }
    __syncthreads();
    for (int i = 0; i < 128; ++i) g += sq[i][d] * sk[i][e];
    if (t < 16) { for (int i = 0; i < 128; ++i) accq += sq[i][t] * sq[i][t]; }
    else if (t < 32) { int dd = t - 16; for (int i = 0; i < 128; ++i) acck += sk[i][dd] * sk[i][dd]; }
    __syncthreads();
  }
  graw[h * 256 + d * 16 + e] = g;
  if (t < 16) sq2[h * 16 + t] = accq;
  else if (t < 32) sk2[h * 16 + t - 16] = acck;
}

__global__ __launch_bounds__(128) void k_ca_softmax(const float* __restrict__ graw, const float* __restrict__ sq2,
                                                    const float* __restrict__ sk2, const float* __restrict__ temp2,
                                                    float* __restrict__ attnCA) {
  int t = threadIdx.x;              // 128 rows: h*16+d
  int h = t >> 4, d = t & 15;
  float sqv = fmaxf(sqrtf(sq2[h * 16 + d]), 1e-12f);
  float tp = temp2[h];
  float lg[16], mx = -3.4e38f;
  for (int e = 0; e < 16; ++e) {
    float skv = fmaxf(sqrtf(sk2[h * 16 + e]), 1e-12f);
    lg[e] = graw[h * 256 + d * 16 + e] / (sqv * skv) * tp;
    mx = fmaxf(mx, lg[e]);
  }
  float sum = 0.f;
  for (int e = 0; e < 16; ++e) { lg[e] = expf(lg[e] - mx); sum += lg[e]; }
  float inv = 1.f / sum;
  for (int e = 0; e < 16; ++e) attnCA[h * 256 + d * 16 + e] = lg[e] * inv;
}

__global__ __launch_bounds__(128) void k_ca_apply(const float* __restrict__ attnCA, const float* __restrict__ qkvv,
                                                  float* __restrict__ xCA) {
  __shared__ float sA[2048];
  int n = blockIdx.x, c = threadIdx.x;
  for (int i = c; i < 2048; i += 128) sA[i] = attnCA[i];
  __syncthreads();
  int h = c >> 4, d = c & 15;
  const float* vr = qkvv + (size_t)n * 512 + 256 + h * 16;
  float s = 0.f;
  for (int e = 0; e < 16; ++e) s += sA[h * 256 + d * 16 + e] * vr[e];
  xCA[(size_t)n * 128 + c] = s;
}

// =====================================================================
// residual / BN / final
// =====================================================================
__global__ __launch_bounds__(256) void k_residual(const float* __restrict__ xs, const float* __restrict__ gamma,
                                                  const float* __restrict__ ySA, const float* __restrict__ yCA,
                                                  float* __restrict__ skipf, __bf16* __restrict__ skipb) {
  int i = blockIdx.x * 256 + threadIdx.x;
  if (i >= NTOK * CCH) return;
  int n = i >> 7, c = i & 127;
  float e = (c < 64) ? ySA[(size_t)n * 64 + c] : yCA[(size_t)n * 64 + (c - 64)];
  float v = xs[i] + gamma[c] * e;
  skipf[i] = v;
  skipb[i] = (__bf16)v;
}

__global__ __launch_bounds__(256) void k_bnstats(const float* __restrict__ t, float* __restrict__ mean,
                                                 float* __restrict__ var) {
  __shared__ float rs[256], rq[256];
  int c = blockIdx.x, t0 = threadIdx.x;
  float s = 0.f, s2 = 0.f;
  for (int n = t0; n < NTOK; n += 256) {
    float v = t[(size_t)n * 128 + c];
    s += v; s2 += v * v;
  }
  rs[t0] = s; rq[t0] = s2; __syncthreads();
  for (int s1 = 128; s1 > 0; s1 >>= 1) {
    if (t0 < s1) { rs[t0] += rs[t0 + s1]; rq[t0] += rq[t0 + s1]; }
    __syncthreads();
  }
  if (t0 == 0) {
    float m = rs[0] * (1.f / NTOK);
    mean[c] = m;
    var[c] = rq[0] * (1.f / NTOK) - m * m;
  }
}

__global__ __launch_bounds__(256) void k_bnapply(const float* __restrict__ t, const float* __restrict__ mean,
                                                 const float* __restrict__ var, const float* __restrict__ w,
                                                 const float* __restrict__ b, const float* __restrict__ skip,
                                                 __bf16* __restrict__ outbf, int addskip) {
  int i = blockIdx.x * 256 + threadIdx.x;
  if (i >= NTOK * CCH) return;
  int c = i & 127;
  float v = (t[i] - mean[c]) * rsqrtf(var[c] + 1e-5f) * w[c] + b[c];
  if (addskip) v += skip[i];
  v = (v > 0.f) ? v : 0.01f * v;
  outbf[i] = (__bf16)v;
}

__global__ __launch_bounds__(256) void k_final(const float* __restrict__ skip, const float* __restrict__ c8o,
                                               float* __restrict__ out) {
  int j = blockIdx.x * 256 + threadIdx.x;         // j = c*NTOK + n  (output layout B,C,H,W,D)
  if (j >= CCH * NTOK) return;
  int c = j / NTOK, n = j % NTOK;
  out[j] = skip[(size_t)n * 128 + c] + c8o[(size_t)n * 128 + c];
}

// =====================================================================
// host launcher
// =====================================================================
extern "C" void kernel_launch(void* const* d_in, const int* in_sizes, int n_in,
                              void* d_out, int out_size, void* d_ws, size_t ws_size,
                              hipStream_t stream) {
  const float* x      = (const float*)d_in[0];
  const int*   rpi    = (const int*)d_in[1];
  const float* rct    = (const float*)d_in[2];
  const float* sls    = (const float*)d_in[3];
  const float* ln_w   = (const float*)d_in[4];
  const float* ln_b   = (const float*)d_in[5];
  const float* gamma  = (const float*)d_in[6];
  const float* qkvv_w = (const float*)d_in[7];
  const float* op_w   = (const float*)d_in[8];
  const float* op_b   = (const float*)d_in[9];
  const float* op2_w  = (const float*)d_in[10];
  const float* op2_b  = (const float*)d_in[11];
  const float* temp   = (const float*)d_in[12];
  const float* temp2  = (const float*)d_in[13];
  const float* sr_w   = (const float*)d_in[14];
  const float* sr_b   = (const float*)d_in[15];
  const float* norm_w = (const float*)d_in[16];
  const float* norm_b = (const float*)d_in[17];
  const float* cpb1_w = (const float*)d_in[18];
  const float* cpb1_b = (const float*)d_in[19];
  const float* cpb2_w = (const float*)d_in[20];
  const float* cpb2_b = (const float*)d_in[21];
  const float* rpb    = (const float*)d_in[22];
  const float* ltok   = (const float*)d_in[23];
  const float* lbias  = (const float*)d_in[24];
  const float* qe     = (const float*)d_in[25];
  const float* kv_w   = (const float*)d_in[26];
  const float* c1_w   = (const float*)d_in[27];
  const float* c1_b   = (const float*)d_in[28];
  const float* bn1_w  = (const float*)d_in[29];
  const float* bn1_b  = (const float*)d_in[30];
  const float* c2_w   = (const float*)d_in[31];
  const float* c2_b   = (const float*)d_in[32];
  const float* bn2_w  = (const float*)d_in[33];
  const float* bn2_b  = (const float*)d_in[34];
  const float* c8_w   = (const float*)d_in[35];
  const float* c8_b   = (const float*)d_in[36];

  char* base = (char*)d_ws;
  size_t off = 0;
  auto alloc = [&](size_t bytes) -> char* {
    char* p = base + off;
    off += (bytes + 255) & ~(size_t)255;
    return p;
  };
  float*  xs     = (float*)alloc((size_t)NTOK * 128 * 4);
  __bf16* lnxs   = (__bf16*)alloc((size_t)NTOK * 128 * 2);
  float*  qkvv   = (float*)alloc((size_t)NTOK * 512 * 4);
  float*  qnorm  = (float*)alloc((size_t)NTOK * 128 * 4);
  float*  srout  = (float*)alloc((size_t)NTOK * 128 * 4);
  float*  xp     = (float*)alloc((size_t)PLN * 128 * 4);
  __bf16* xpln   = (__bf16*)alloc((size_t)PLN * 128 * 2);
  float*  kvp    = (float*)alloc((size_t)PLN * 256 * 4);
  float*  kpooln = (float*)alloc((size_t)NHEADS * PLN * 16 * 4);
  float*  vpool  = (float*)alloc((size_t)NHEADS * PLN * 16 * 4);
  float*  tbl    = (float*)alloc((size_t)TTBL * 8 * 4);
  float*  graw   = (float*)alloc((size_t)NHEADS * 256 * 4);
  float*  sq2    = (float*)alloc(128 * 4);
  float*  sk2    = (float*)alloc(128 * 4);
  float*  attnCA = (float*)alloc((size_t)NHEADS * 256 * 4);
  float*  xSA    = (float*)alloc((size_t)NTOK * 128 * 4);
  float*  xCA    = (float*)alloc((size_t)NTOK * 128 * 4);
  __bf16* xSAb   = (__bf16*)alloc((size_t)NTOK * 128 * 2);
  __bf16* xCAb   = (__bf16*)alloc((size_t)NTOK * 128 * 2);
  float*  ySA    = (float*)alloc((size_t)NTOK * 64 * 4);
  float*  yCA    = (float*)alloc((size_t)NTOK * 64 * 4);
  float*  skipf  = (float*)alloc((size_t)NTOK * 128 * 4);
  __bf16* skipb  = (__bf16*)alloc((size_t)NTOK * 128 * 2);
  float*  t1     = (float*)alloc((size_t)NTOK * 128 * 4);
  __bf16* a1b    = (__bf16*)alloc((size_t)NTOK * 128 * 2);
  float*  t2     = (float*)alloc((size_t)NTOK * 128 * 4);
  __bf16* a2b    = (__bf16*)alloc((size_t)NTOK * 128 * 2);
  float*  c8o    = (float*)alloc((size_t)NTOK * 128 * 4);
  float*  m1     = (float*)alloc(128 * 4);
  float*  v1     = (float*)alloc(128 * 4);
  float*  m2     = (float*)alloc(128 * 4);
  float*  v2     = (float*)alloc(128 * 4);
  __bf16* qkvvwb = (__bf16*)alloc((size_t)512 * 128 * 2);
  __bf16* srwb   = (__bf16*)alloc((size_t)128 * 128 * 2);
  __bf16* kvwb   = (__bf16*)alloc((size_t)256 * 128 * 2);
  __bf16* opwb   = (__bf16*)alloc((size_t)64 * 128 * 2);
  __bf16* op2wb  = (__bf16*)alloc((size_t)64 * 128 * 2);
  __bf16* c8wb   = (__bf16*)alloc((size_t)128 * 128 * 2);
  __bf16* wpk1   = (__bf16*)alloc((size_t)27 * 128 * 128 * 2);
  __bf16* wpk2   = (__bf16*)alloc((size_t)27 * 128 * 128 * 2);

  const int NC = NTOK * CCH;            // 1,179,648
  const int EB = (NC + 255) / 256;      // 4608

  // ---- prep ----
  k_transpose_in<<<EB, 256, 0, stream>>>(x, xs);
  k_layernorm<<<NTOK, 128, 0, stream>>>(xs, ln_w, ln_b, lnxs);
  k_f2bf<<<(512 * 128 + 255) / 256, 256, 0, stream>>>(qkvv_w, qkvvwb, 512 * 128);
  k_f2bf<<<(128 * 128 + 255) / 256, 256, 0, stream>>>(sr_w, srwb, 128 * 128);
  k_f2bf<<<(256 * 128 + 255) / 256, 256, 0, stream>>>(kv_w, kvwb, 256 * 128);
  k_f2bf<<<(64 * 128 + 255) / 256, 256, 0, stream>>>(op_w, opwb, 64 * 128);
  k_f2bf<<<(64 * 128 + 255) / 256, 256, 0, stream>>>(op2_w, op2wb, 64 * 128);
  k_f2bf<<<(128 * 128 + 255) / 256, 256, 0, stream>>>(c8_w, c8wb, 128 * 128);
  k_pack_convw<<<(27 * 128 * 128 + 255) / 256, 256, 0, stream>>>(c1_w, wpk1);
  k_pack_convw<<<(27 * 128 * 128 + 255) / 256, 256, 0, stream>>>(c2_w, wpk2);

  // ---- qkvv GEMM (WMMA), then channel-attn Gram on RAW q/k, then q/k norm ----
  k_gemm128<<<576, 256, 0, stream>>>(lnxs, qkvvwb, nullptr, qkvv, 32, 0);
  k_ca_gram<<<NHEADS, 256, 0, stream>>>(qkvv, graw, sq2, sk2);
  k_ca_softmax<<<1, 128, 0, stream>>>(graw, sq2, sk2, temp2, attnCA);
  k_qk_norm<<<(NTOK * NHEADS + 255) / 256, 256, 0, stream>>>(qkvv, qnorm, qe, temp, sls);

  // ---- pooled branch ----
  k_gemm128<<<576, 256, 0, stream>>>(lnxs, srwb, sr_b, srout, 8, 1);  // gelu
  k_pool<<<PLN, 128, 0, stream>>>(srout, xp);
  k_layernorm<<<PLN, 128, 0, stream>>>(xp, norm_w, norm_b, xpln);
  k_gemm128<<<9, 256, 0, stream>>>(xpln, kvwb, nullptr, kvp, 16, 0);
  k_norm_kpool<<<(NHEADS * PLN + 255) / 256, 256, 0, stream>>>(kvp, kpooln, vpool);
  k_cpb<<<(TTBL + 255) / 256, 256, 0, stream>>>(rct, cpb1_w, cpb1_b, cpb2_w, cpb2_b, tbl);

  // ---- fused local+pool attention, channel attention apply ----
  k_attn<<<dim3(NTOK / 64, NHEADS), 64, 0, stream>>>(qkvv, qnorm, kpooln, vpool, tbl, rpi, rpb, ltok, lbias, xSA);
  k_ca_apply<<<NTOK, 128, 0, stream>>>(attnCA, qkvv, xCA);

  // ---- output projections + residual ----
  k_f2bf<<<EB, 256, 0, stream>>>(xSA, xSAb, NC);
  k_f2bf<<<EB, 256, 0, stream>>>(xCA, xCAb, NC);
  k_gemm128<<<576, 256, 0, stream>>>(xSAb, opwb, op_b, ySA, 4, 0);
  k_gemm128<<<576, 256, 0, stream>>>(xCAb, op2wb, op2_b, yCA, 4, 0);
  k_residual<<<EB, 256, 0, stream>>>(xs, gamma, ySA, yCA, skipf, skipb);

  // ---- conv block (implicit-GEMM WMMA, async-LDS staged activations) ----
  k_conv3d_wmma<<<576, 256, 0, stream>>>(skipb, wpk1, c1_b, t1);
  k_bnstats<<<128, 256, 0, stream>>>(t1, m1, v1);
  k_bnapply<<<EB, 256, 0, stream>>>(t1, m1, v1, bn1_w, bn1_b, nullptr, a1b, 0);
  k_conv3d_wmma<<<576, 256, 0, stream>>>(a1b, wpk2, c2_b, t2);
  k_bnstats<<<128, 256, 0, stream>>>(t2, m2, v2);
  k_bnapply<<<EB, 256, 0, stream>>>(t2, m2, v2, bn2_w, bn2_b, skipf, a2b, 1);
  k_gemm128<<<576, 256, 0, stream>>>(a2b, c8wb, c8_b, c8o, 8, 0);

  // ---- final residual + layout transpose ----
  k_final<<<EB, 256, 0, stream>>>(skipf, c8o, (float*)d_out);
}
